// CausalSelfAttention_36679020708023
// MI455X (gfx1250) — matrix operand order repaired and checked by reading, not compiled, not run on hardware
//
#include <hip/hip_runtime.h>

// ---------------------------------------------------------------------------
// CDNA5 (gfx1250) causal self-attention:
//   qkv GEMM (WMMA bf16) -> RoPE -> flash attention (WMMA bf16, TDM tile
//   loads via tensor_load_to_lds + s_wait_tensorcnt) -> out GEMM (WMMA bf16)
// ---------------------------------------------------------------------------

typedef __attribute__((ext_vector_type(16))) __bf16 v16bf;
typedef __attribute__((ext_vector_type(8)))  float  v8f;
typedef __attribute__((ext_vector_type(4)))  unsigned int u32x4;
typedef __attribute__((ext_vector_type(8)))  unsigned int u32x8;

#define WMMA_BF16(a, b, c) \
    __builtin_amdgcn_wmma_f32_16x16x32_bf16(false, (a), false, (b), (short)0, (c), false, false)

#define BB 2
#define TT 2048
#define CC 2048
#define HH 16
#define DD 128

__device__ inline unsigned int pack_bf16(float a, float b) {
    union { __bf16 h[2]; unsigned int u; } p;
    p.h[0] = (__bf16)a; p.h[1] = (__bf16)b;
    return p.u;
}

// ---------------------------------------------------------------------------
// TDM: 2-D bf16 tile load, global -> LDS (row-major into LDS, contiguous).
// D# layout per CDNA5 ISA. One issue per call
// (TDM ignores EXEC). lds_off = LDS byte offset (= low 32 bits of flat addr).
// ---------------------------------------------------------------------------
__device__ inline void tdm_load_tile_2d(unsigned int lds_off, const void* gaddr,
                                        unsigned int tile_d0, unsigned int tile_d1,
                                        unsigned int ten_d0, unsigned int ten_d1,
                                        unsigned int stride0_elems)
{
    unsigned long long ga = (unsigned long long)gaddr;
    u32x4 g0;
    g0[0] = 1u;                                            // count=1, user D#
    g0[1] = lds_off;                                       // lds_addr (bytes)
    g0[2] = (unsigned int)(ga & 0xFFFFFFFFu);              // global_addr[31:0]
    g0[3] = (unsigned int)((ga >> 32) & 0x01FFFFFFu)       // global_addr[56:32]
          | (2u << 30);                                    // type = 2 (image)
    u32x8 g1;
    g1[0] = (1u << 16);                                    // data_size=1 (2B)
    g1[1] = (ten_d0 & 0xFFFFu) << 16;                      // tensor_dim0[15:0]
    g1[2] = ((ten_d0 >> 16) & 0xFFFFu) | ((ten_d1 & 0xFFFFu) << 16);
    g1[3] = ((ten_d1 >> 16) & 0xFFFFu) | ((tile_d0 & 0xFFFFu) << 16);
    g1[4] = (tile_d1 & 0xFFFFu);                           // tile_dim1, tile_dim2=0
    g1[5] = stride0_elems;                                 // tensor_dim0_stride lo32
    g1[6] = 0u;                                            // stride0 hi / dim1_stride lo
    g1[7] = 0u;
    asm volatile("tensor_load_to_lds %0, %1" :: "s"(g0), "s"(g1) : "memory");
}

// --------------------------------------------------------------------------
// Kernel 1: qkv = x @ w_qkv (M=4096, K=2048, N=6144) -> Q,K [B,H,T,D] bf16,
// V transposed [B,H,D,T] bf16 (so attention V tiles land LDS-vector-friendly).
// Block: 256 thr = 8 waves (4 m x 2 n), tile 64(M) x 128(N).
// LDS: A tile 64x32 row-major, B tile transposed [128 col][32 k].
// --------------------------------------------------------------------------
__global__ __launch_bounds__(256) void qkv_gemm_kernel(
    const float* __restrict__ x, const float* __restrict__ wqkv,
    __bf16* __restrict__ Q, __bf16* __restrict__ K, __bf16* __restrict__ Vt)
{
    __shared__ __align__(16) __bf16 sA[64 * 32];
    __shared__ __align__(16) __bf16 sBt[128 * 32];

    const int tid  = threadIdx.x;
    const int lane = tid & 31;
    const int wave = tid >> 5;
    const int wm   = wave >> 1;
    const int wn   = wave & 1;
    const int nl   = lane & 15;
    const int hl   = lane >> 4;
    const int rowBase = blockIdx.y * 64;
    const int colBase = blockIdx.x * 128;

    const int bc = tid & 127;          // B-tile column this thread owns
    const int bk = tid >> 7;           // 0/1 -> k half

    v8f acc[4] = {};

    for (int k0 = 0; k0 < CC; k0 += 32) {
        __syncthreads();
        // A tile 64x32: float2 -> packed bf16 pair -> one b32 LDS store
        #pragma unroll
        for (int e = 0; e < 4; ++e) {
            int pli = e * 256 + tid;             // pair index, row = pli>>4
            int r = pli >> 4, cp = pli & 15;
            float2 f = *(const float2*)&x[(size_t)(rowBase + r) * CC + k0 + 2 * cp];
            ((unsigned int*)sA)[pli] = pack_bf16(f.x, f.y);
        }
        // B tile transposed: thread owns column bc, k-half bk (16 k values)
        {
            unsigned int* dst = (unsigned int*)&sBt[bc * 32 + bk * 16];
            #pragma unroll
            for (int p = 0; p < 8; ++p) {
                float f0 = wqkv[(size_t)(k0 + bk * 16 + 2 * p    ) * (3 * CC) + colBase + bc];
                float f1 = wqkv[(size_t)(k0 + bk * 16 + 2 * p + 1) * (3 * CC) + colBase + bc];
                dst[p] = pack_bf16(f0, f1);
            }
        }
        __syncthreads();

        v16bf a;
        #pragma unroll
        for (int i = 0; i < 16; ++i) {
            int kk = i + (hl << 4);
            a[i] = sA[(wm * 16 + nl) * 32 + kk];
        }
        #pragma unroll
        for (int nt = 0; nt < 4; ++nt) {
            int col = wn * 64 + nt * 16 + nl;
            v16bf b;
            #pragma unroll
            for (int i = 0; i < 16; ++i)          // contiguous -> ds_load_b128 x2
                b[i] = sBt[col * 32 + hl * 16 + i];
            acc[nt] = WMMA_BF16(a, b, acc[nt]);
        }
    }

    #pragma unroll
    for (int nt = 0; nt < 4; ++nt) {
        int cn     = colBase + wn * 64 + nt * 16 + nl;
        int sel    = cn >> 11;                    // 0=q 1=k 2=v
        int within = cn & 2047;
        int h      = within >> 7;
        int d      = within & 127;
        __bf16* dst = (sel == 0) ? Q : (sel == 1) ? K : Vt;
        #pragma unroll
        for (int j = 0; j < 8; ++j) {
            int r = rowBase + wm * 16 + j + 8 * hl;
            int b = r >> 11, t = r & 2047;
            size_t idx = (sel == 2)
                       ? (((size_t)b * HH + h) * DD + d) * TT + t    // V: [B,H,D,T]
                       : (((size_t)b * HH + h) * TT + t) * DD + d;   // Q,K: [B,H,T,D]
            dst[idx] = (__bf16)acc[nt][j];
        }
    }
}

// --------------------------------------------------------------------------
// Kernel 2: RoPE on Q (1/sqrt(D) folded) and K, in place.
// --------------------------------------------------------------------------
__global__ __launch_bounds__(256) void rope_kernel(__bf16* __restrict__ Q,
                                                   __bf16* __restrict__ K)
{
    int idx = blockIdx.x * 256 + threadIdx.x;     // < 2*16*2048*64
    int d2 = idx & 63;
    int t  = (idx >> 6) & 2047;
    int bh = idx >> 17;
    size_t base = ((size_t)bh * TT + t) * DD;

    float inv = __expf(-0.14391156f * (float)d2); // 10000^(-d2/64)
    float f = (float)t * inv;
    float c = __cosf(f), s = __sinf(f);

    const float sc = 0.08838834764831845f;        // 1/sqrt(128)
    float q1 = (float)Q[base + d2], q2 = (float)Q[base + d2 + 64];
    Q[base + d2]      = (__bf16)((q1 * c - q2 * s) * sc);
    Q[base + d2 + 64] = (__bf16)((q2 * c + q1 * s) * sc);
    float k1 = (float)K[base + d2], k2 = (float)K[base + d2 + 64];
    K[base + d2]      = (__bf16)(k1 * c - k2 * s);
    K[base + d2 + 64] = (__bf16)(k2 * c + k1 * s);
}

// --------------------------------------------------------------------------
// Kernel 3: causal flash attention per (b,h). Block = 128 query rows, 8 waves.
// K tile (64x128, row-major) and Vt tile (128 d x 64 t) DMA'd by the TDM.
// LDS: K 16KB | Vt 16KB | P 16KB (per-wave) = 48KB.
// --------------------------------------------------------------------------
__global__ __launch_bounds__(256) void attn_kernel(
    const __bf16* __restrict__ Q, const __bf16* __restrict__ K,
    const __bf16* __restrict__ Vt, __bf16* __restrict__ Y)
{
    __shared__ __align__(16) __bf16 smem[24576];   // 48 KB
    __bf16* sK  = smem;                            // [64 key][128 d]
    __bf16* sVt = smem + 8192;                     // [128 d][64 key]
    __bf16* sP  = smem + 16384;                    // 8 waves * [16][64]

    const int tid  = threadIdx.x;
    const int lane = tid & 31;
    const int wave = tid >> 5;
    const int nl   = lane & 15;
    const int hl   = lane >> 4;
    const int bh   = blockIdx.y;
    const int b    = bh >> 4, h = bh & 15;
    const int q0   = blockIdx.x * 128;

    const __bf16* Qh  = Q  + (size_t)bh * TT * DD;
    const __bf16* Kh  = K  + (size_t)bh * TT * DD;
    const __bf16* Vth = Vt + (size_t)bh * DD * TT;

    // flat->LDS: LDS byte offset = low 32 bits of generic address (ISA §10.2)
    const unsigned int ldsK = (unsigned int)(uintptr_t)sK;
    const unsigned int ldsV = (unsigned int)(uintptr_t)sVt;

    // Preload Q fragments: 2 x b128 per 32-d chunk per lane
    v16bf qf[4];
    {
        int t = q0 + wave * 16 + nl;
        const uint4* qrow = (const uint4*)(Qh + (size_t)t * DD);
        #pragma unroll
        for (int dc = 0; dc < 4; ++dc) {
            union { uint4 u[2]; v16bf v; } u;
            u.u[0] = qrow[dc * 4 + hl];        // d = dc*32 + hl*8 .. +8
            u.u[1] = qrow[dc * 4 + 2 + hl];    // d = dc*32 + 16 + hl*8 .. +8
            qf[dc] = u.v;
        }
    }

    v8f o[8] = {};
    float mrow[8], lrow[8];
    #pragma unroll
    for (int j = 0; j < 8; ++j) { mrow[j] = -1e30f; lrow[j] = 0.0f; }

    const int nKB = blockIdx.x * 2 + 2;            // causal pruning
    for (int kb = 0; kb < nKB; ++kb) {
        __syncthreads();                           // LDS consumers done
        if (wave == 0) {
            // K tile: 64 rows x 128 d of [T- kb*64, 128] tensor, stride 128
            tdm_load_tile_2d(ldsK, Kh + (size_t)kb * 64 * DD,
                             128u, 64u, 128u, (unsigned)(TT - kb * 64), 128u);
            // Vt tile: 128 d-rows x 64 t of [128, T] tensor, stride T
            tdm_load_tile_2d(ldsV, Vth + (size_t)kb * 64,
                             64u, 128u, (unsigned)(TT - kb * 64), 128u, (unsigned)TT);
            __builtin_amdgcn_s_wait_tensorcnt(0);
        }
        __syncthreads();                           // tiles visible to all waves

        // S = Q @ K^T  (16 rows x 64 keys per wave)
        v8f s[4] = {};
        #pragma unroll
        for (int nt = 0; nt < 4; ++nt) {
            #pragma unroll
            for (int dc = 0; dc < 4; ++dc) {
                v16bf bfrag;
                #pragma unroll
                for (int i = 0; i < 16; ++i)       // table order (2 runs of 8) -> b128 x2
                    bfrag[i] = sK[(nt * 16 + nl) * DD + dc * 32 + (i & 7) + ((i >> 3) << 4) + (hl << 3)];
                s[nt] = WMMA_BF16(qf[dc], bfrag, s[nt]);
            }
        }

        // causal mask + online softmax (rows m=j+8*hl, cols n=nt*16+nl)
        float alpha[8];
        #pragma unroll
        for (int j = 0; j < 8; ++j) {
            int qi = q0 + wave * 16 + j + 8 * hl;
            float mx = -1e30f;
            #pragma unroll
            for (int nt = 0; nt < 4; ++nt) {
                int kj = kb * 64 + nt * 16 + nl;
                float vS = (kj <= qi) ? s[nt][j] : -1e30f;
                s[nt][j] = vS;
                mx = fmaxf(mx, vS);
            }
            #pragma unroll
            for (int off = 1; off < 16; off <<= 1)
                mx = fmaxf(mx, __shfl_xor(mx, off, 32));
            float mnew = fmaxf(mrow[j], mx);
            float al   = __expf(mrow[j] - mnew);
            float rs   = 0.0f;
            #pragma unroll
            for (int nt = 0; nt < 4; ++nt) {
                float p = __expf(s[nt][j] - mnew);
                s[nt][j] = p;
                rs += p;
            }
            #pragma unroll
            for (int off = 1; off < 16; off <<= 1)
                rs += __shfl_xor(rs, off, 32);
            lrow[j] = lrow[j] * al + rs;
            mrow[j] = mnew;
            alpha[j] = al;
        }

        #pragma unroll
        for (int nd = 0; nd < 8; ++nd)
            #pragma unroll
            for (int j = 0; j < 8; ++j)
                o[nd][j] *= alpha[j];

        // P: C-layout -> A-layout via per-wave LDS region
        __bf16* sp = sP + wave * 1024;
        #pragma unroll
        for (int nt = 0; nt < 4; ++nt)
            #pragma unroll
            for (int j = 0; j < 8; ++j)
                sp[(j + 8 * hl) * 64 + nt * 16 + nl] = (__bf16)s[nt][j];
        __syncthreads();

        // O += P @ V  (V fragments contiguous from transposed LDS tile)
        #pragma unroll
        for (int c = 0; c < 2; ++c) {
            v16bf pa;
            #pragma unroll
            for (int i = 0; i < 16; ++i) {
                int kk = i + (hl << 4);
                pa[i] = sp[nl * 64 + c * 32 + kk];
            }
            #pragma unroll
            for (int nd = 0; nd < 8; ++nd) {
                v16bf vb;
                #pragma unroll
                for (int i = 0; i < 16; ++i)       // contiguous -> b128 x2
                    vb[i] = sVt[(nd * 16 + nl) * 64 + c * 32 + hl * 16 + i];
                o[nd] = WMMA_BF16(pa, vb, o[nd]);
            }
        }
    }

    // Normalize, store Y as [B,T,H,D] bf16 (== [B*T, C] rows for out GEMM)
    #pragma unroll
    for (int nd = 0; nd < 8; ++nd)
        #pragma unroll
        for (int j = 0; j < 8; ++j) {
            int m = j + 8 * hl;
            int t = q0 + wave * 16 + m;
            float val = o[nd][j] / lrow[j];
            Y[(((size_t)b * TT + t) * HH + h) * DD + nd * 16 + nl] = (__bf16)val;
        }
}

// --------------------------------------------------------------------------
// Kernel 4: out = Y @ w_out (M=4096, K=2048, N=2048), f32 output.
// Same structure as kernel 1; A already bf16, B transposed in LDS.
// --------------------------------------------------------------------------
__global__ __launch_bounds__(256) void out_gemm_kernel(
    const __bf16* __restrict__ Yg, const float* __restrict__ wout,
    float* __restrict__ out)
{
    __shared__ __align__(16) __bf16 sA[64 * 32];
    __shared__ __align__(16) __bf16 sBt[128 * 32];

    const int tid  = threadIdx.x;
    const int lane = tid & 31;
    const int wave = tid >> 5;
    const int wm   = wave >> 1;
    const int wn   = wave & 1;
    const int nl   = lane & 15;
    const int hl   = lane >> 4;
    const int rowBase = blockIdx.y * 64;
    const int colBase = blockIdx.x * 128;
    const int bc = tid & 127;
    const int bk = tid >> 7;

    v8f acc[4] = {};

    for (int k0 = 0; k0 < CC; k0 += 32) {
        __syncthreads();
        #pragma unroll
        for (int e = 0; e < 4; ++e) {              // A tile: b32 copies
            int li = e * 256 + tid;
            int r = li >> 4, cp = li & 15;
            ((unsigned int*)sA)[li] =
                ((const unsigned int*)Yg)[(size_t)(rowBase + r) * (CC / 2) + (k0 >> 1) + cp];
        }
        {
            unsigned int* dst = (unsigned int*)&sBt[bc * 32 + bk * 16];
            #pragma unroll
            for (int p = 0; p < 8; ++p) {
                float f0 = wout[(size_t)(k0 + bk * 16 + 2 * p    ) * CC + colBase + bc];
                float f1 = wout[(size_t)(k0 + bk * 16 + 2 * p + 1) * CC + colBase + bc];
                dst[p] = pack_bf16(f0, f1);
            }
        }
        __syncthreads();

        v16bf a;
        #pragma unroll
        for (int i = 0; i < 16; ++i) {
            int kk = i + (hl << 4);
            a[i] = sA[(wm * 16 + nl) * 32 + kk];
        }
        #pragma unroll
        for (int nt = 0; nt < 4; ++nt) {
            int col = wn * 64 + nt * 16 + nl;
            v16bf b;
            #pragma unroll
            for (int i = 0; i < 16; ++i)
                b[i] = sBt[col * 32 + hl * 16 + i];
            acc[nt] = WMMA_BF16(a, b, acc[nt]);
        }
    }

    #pragma unroll
    for (int nt = 0; nt < 4; ++nt) {
        int cn = colBase + wn * 64 + nt * 16 + nl;
        #pragma unroll
        for (int j = 0; j < 8; ++j) {
            int r = rowBase + wm * 16 + j + 8 * hl;
            out[(size_t)r * CC + cn] = acc[nt][j];
        }
    }
}

// --------------------------------------------------------------------------
extern "C" void kernel_launch(void* const* d_in, const int* in_sizes, int n_in,
                              void* d_out, int out_size, void* d_ws, size_t ws_size,
                              hipStream_t stream)
{
    const float* x    = (const float*)d_in[0];   // [B,T,C]
    const float* wqkv = (const float*)d_in[1];   // [C,3C]
    const float* wout = (const float*)d_in[2];   // [C,C]
    float* out = (float*)d_out;                  // [B,T,C] f32

    char* ws = (char*)d_ws;
    __bf16* Q  = (__bf16*)(ws);                             // [B,H,T,D] 16MB
    __bf16* K  = (__bf16*)(ws + (size_t)16 * 1024 * 1024);  // [B,H,T,D] 16MB
    __bf16* Vt = (__bf16*)(ws + (size_t)32 * 1024 * 1024);  // [B,H,D,T] 16MB
    __bf16* Y  = (__bf16*)(ws + (size_t)48 * 1024 * 1024);  // [B,T,H,D] 16MB

    qkv_gemm_kernel<<<dim3(48, 64), 256, 0, stream>>>(x, wqkv, Q, K, Vt);
    rope_kernel<<<16384, 256, 0, stream>>>(Q, K);
    attn_kernel<<<dim3(16, 32), 256, 0, stream>>>(Q, K, Vt, Y);
    out_gemm_kernel<<<dim3(16, 64), 256, 0, stream>>>(Y, wout, out);
}
